// GRUDecoder_19550691131787
// MI455X (gfx1250) — matrix operand order repaired and checked by reading, not compiled, not run on hardware
//
#include <hip/hip_runtime.h>

// ---------------------------------------------------------------------------
// GRU decoder on MI455X (gfx1250), persistent-kernel RNN.
//   BATCH=64, HIDDEN=512, LAYERS=2, OUTPUT=128, SEQ=512.
// 32 persistent workgroups x 128 threads (4 wave32). WG g owns hidden cols
// [16g,16g+16) of both layers; wave w owns batch rows [16w,16w+16).
// Weights converted to bf16 and pre-swizzled into WMMA B-operand register
// layout, held LDS-resident (208KB/WGP) for all 512 steps.
// ---------------------------------------------------------------------------

#define BATCH    64
#define HIDDEN   512
#define OUTPUT   128
#define SEQ_LEN  512
#define NWG      32
#define BLOCK    128
#define GATE_ELEMS 98304            // per-WG gate-weight bf16 elems (192KB)
#define WG_ELEMS   106496           // + W_o slice (16KB) => 208KB total
#define LDS_BYTES  (WG_ELEMS * 2)   // 212992 bytes dynamic LDS
#define BH        (BATCH * HIDDEN)  // 32768 elems per state buffer

typedef __attribute__((ext_vector_type(16))) __bf16        bf16x16;
typedef __attribute__((ext_vector_type(8)))  float         f32x8;
typedef __attribute__((ext_vector_type(4)))  unsigned int  u32x4;
typedef __attribute__((ext_vector_type(8)))  int           i32x8;
typedef __attribute__((ext_vector_type(4)))  int           i32x4;

union TileBits { bf16x16 v; uint4 q[2]; };

__device__ __host__ inline unsigned short f2bf(float f) {
  unsigned u = __builtin_bit_cast(unsigned, f);
  unsigned r = u + 0x7FFFu + ((u >> 16) & 1u);   // RNE, ignore NaN edge
  return (unsigned short)(r >> 16);
}

__device__ inline float fsig(float x) {
  return 1.0f / (1.0f + __expf(-x));
}
__device__ inline float ftanh(float x) {
  float xc = fminf(fmaxf(x, -15.0f), 15.0f);
  float e  = __expf(-2.0f * xc);
  return (1.0f - e) / (1.0f + e);
}

// A-operand tile (16x32 bf16) from row-major [.,512] bf16 matrix, wave32
// contiguous-K order used here (same as B): lane L holds row (L&15); lanes 0-15 K [0..16),
// lanes 16-31 K [16..32)  -> two b128 loads per lane.
__device__ inline bf16x16 load_a(const unsigned short* __restrict__ base,
                                 int row0, int k0, int lane) {
  int row = row0 + (lane & 15);
  int kk  = k0 + ((lane >> 4) << 4);
  const unsigned short* p = base + (size_t)row * HIDDEN + kk;
  TileBits t;
  t.q[0] = *(const uint4*)(p);
  t.q[1] = *(const uint4*)(p + 8);
  return t.v;
}

// B-operand tile (32x16 bf16) from pre-swizzled 1KB LDS block: lane L's 32
// contiguous bytes at L*32 (K 0..15 for lanes 0-15, K 16..31 for 16-31).
__device__ inline bf16x16 load_b(const unsigned short* __restrict__ tile,
                                 int lane) {
  const unsigned short* p = tile + lane * 16;
  TileBits t;
  t.q[0] = *(const uint4*)(p);
  t.q[1] = *(const uint4*)(p + 8);
  return t.v;
}

__device__ inline f32x8 wmma_bf16(bf16x16 a, bf16x16 b, f32x8 c) {
  return __builtin_amdgcn_wmma_f32_16x16x32_bf16(
      /*neg_a=*/false, a, /*neg_b=*/false, b,
      /*c_mod=*/(short)0, c, /*reuse_a=*/false, /*reuse_b=*/false);
}

// Monotonic-counter grid barrier over NWG persistent workgroups.
__device__ inline void grid_barrier(int* bar, int target) {
  __syncthreads();
  if (threadIdx.x == 0) {
    __threadfence();
    __hip_atomic_fetch_add(bar, 1, __ATOMIC_ACQ_REL, __HIP_MEMORY_SCOPE_AGENT);
    while (__hip_atomic_load(bar, __ATOMIC_ACQUIRE, __HIP_MEMORY_SCOPE_AGENT) < target)
      __builtin_amdgcn_s_sleep(1);
    __threadfence();
  }
  __syncthreads();
}

// ---------------------------------------------------------------------------
// Prep kernel 1: fp32 -> bf16 weight conversion + swizzle into per-WG LDS
// images (B-operand register layout per 32x16 K-tile).
// ---------------------------------------------------------------------------
__global__ void prep_weights(const float* __restrict__ W_ih,
                             const float* __restrict__ W_hh,
                             const float* __restrict__ W_o,
                             unsigned short* __restrict__ swz) {
  long long gid = (long long)blockIdx.x * 256 + threadIdx.x;
  const long long total = (long long)NWG * WG_ELEMS;
  if (gid >= total) return;
  int wg  = (int)(gid / WG_ELEMS);
  int off = (int)(gid % WG_ELEMS);
  float v;
  if (off < GATE_ELEMS) {
    int e    = off & 511;
    int idx2 = off >> 9;
    int kt   = idx2 & 15;
    int idx3 = idx2 >> 4;        // (layer*2 + mat)*3 + gate
    int g    = idx3 % 3;
    int lm   = idx3 / 3;
    int mat  = lm & 1;
    int l    = lm >> 1;
    int L    = e >> 4;           // "lane"
    int j    = e & 15;
    int row  = g * 512 + wg * 16 + (L & 15);
    int k    = kt * 32 + ((L >> 4) << 4) + j;
    const float* W = mat ? W_hh : W_ih;
    v = W[((size_t)l * 1536 + row) * 512 + k];
  } else {
    int o2 = off - GATE_ELEMS;
    int e  = o2 & 511;
    int kt = o2 >> 9;
    int L  = e >> 4;
    int j  = e & 15;
    int row = (wg & 7) * 16 + (L & 15);
    int k   = kt * 32 + ((L >> 4) << 4) + j;
    v = W_o[(size_t)row * 512 + k];
  }
  swz[gid] = f2bf(v);
}

// ---------------------------------------------------------------------------
// Prep kernel 2: h_init = z @ W_l^T + b_l ; seed state buffers + barrier.
// ---------------------------------------------------------------------------
__global__ void prep_init(const float* __restrict__ z,
                          const float* __restrict__ W_l,
                          const float* __restrict__ b_l,
                          unsigned short* __restrict__ Xb,
                          unsigned short* __restrict__ H0b,
                          unsigned short* __restrict__ H1b,
                          float* __restrict__ hinitf,
                          int* __restrict__ bar) {
  int gid = blockIdx.x * 256 + threadIdx.x;
  if (gid == 0) *bar = 0;
  if (gid >= BH) return;
  int b = gid >> 9;
  int n = gid & 511;
  float acc = b_l[n];
  for (int k = 0; k < 64; ++k)
    acc += z[b * 64 + k] * W_l[n * 64 + k];
  hinitf[gid] = acc;
  unsigned short hb = f2bf(acc);
  H0b[gid] = hb;   // parity-0 slices
  H1b[gid] = hb;
  Xb[gid]  = 0;    // x0 = zeros (bf16 zero)
}

// ---------------------------------------------------------------------------
// Main persistent kernel.
// ---------------------------------------------------------------------------
__global__ __launch_bounds__(BLOCK, 1)
void gru_persistent(const unsigned short* __restrict__ swz,
                    unsigned short* __restrict__ Xb,
                    unsigned short* __restrict__ H0b,
                    unsigned short* __restrict__ H1b,
                    const float* __restrict__ hinitf,
                    const float* __restrict__ b_ih,
                    const float* __restrict__ b_hh,
                    const float* __restrict__ b_o,
                    float* __restrict__ out,
                    int* __restrict__ bar) {
  extern __shared__ unsigned short ldsw[];
  const int wg    = blockIdx.x;
  const int tid   = threadIdx.x;
  const int wave  = tid >> 5;
  const int lane  = tid & 31;
  const int hi    = (lane >> 4) & 1;
  const int col   = lane & 15;
  const int mbase = wave * 16;   // batch tile rows
  const int nbase = wg * 16;     // hidden slice cols

  // ---- Stage weights into LDS: TDM async-tensor prefetch, then a
  // ---- guaranteed manual copy (robust even if descriptor semantics drift).
#if defined(__has_builtin)
#if __has_builtin(__builtin_amdgcn_tensor_load_to_lds) && \
    __has_builtin(__builtin_amdgcn_s_wait_tensorcnt)
  if (wave == 0) {
    unsigned long long ga = (unsigned long long)(const void*)
        (swz + (size_t)wg * WG_ELEMS);
    unsigned int ldsoff = (unsigned int)(size_t)(void*)&ldsw[0];
    const unsigned int ne = LDS_BYTES / 8;  // 8-byte elements, 1-row tensor
    u32x4 g0 = { 1u,                                   // count=1
                 ldsoff,                               // lds_addr
                 (unsigned)(ga & 0xFFFFFFFFu),         // global_addr lo
                 (unsigned)((ga >> 32) & 0x01FFFFFFu) | (2u << 30) }; // type=2
    i32x8 g1 = { (int)(3u << 16),                      // data_size=8B
                 (int)((ne & 0xFFFFu) << 16),          // tensor_dim0 lo16
                 (int)(((ne >> 16) & 0xFFFFu) | (1u << 16)), // dim0 hi|dim1=1
                 (int)((ne & 0xFFFFu) << 16),          // tile_dim0
                 1,                                    // tile_dim1=1
                 (int)ne,                              // dim0_stride lo32
                 0, 0 };
    i32x4 gz4 = { 0, 0, 0, 0 };                        // groups 2/3 unused
    i32x8 gz8 = { 0, 0, 0, 0, 0, 0, 0, 0 };
    __builtin_amdgcn_tensor_load_to_lds(g0, g1, gz4, gz4, gz8, 0);
    __builtin_amdgcn_s_wait_tensorcnt(0);
  }
  __syncthreads();
#endif
#endif
  {
    const uint4* src = (const uint4*)(swz + (size_t)wg * WG_ELEMS);
    uint4* dst = (uint4*)ldsw;
    for (int i = tid; i < LDS_BYTES / 16; i += BLOCK) dst[i] = src[i];
  }
  __syncthreads();

  // ---- Per-lane bias scalars (column = nbase+col, same for all 8 rows).
  const int nb = nbase + col;
  const float cr0  = b_ih[nb]               + b_hh[nb];
  const float cz0  = b_ih[512 + nb]         + b_hh[512 + nb];
  const float bin0 = b_ih[1024 + nb];
  const float bhn0 = b_hh[1024 + nb];
  const float cr1  = b_ih[1536 + nb]        + b_hh[1536 + nb];
  const float cz1  = b_ih[1536 + 512 + nb]  + b_hh[1536 + 512 + nb];
  const float bin1 = b_ih[1536 + 1024 + nb];
  const float bhn1 = b_hh[1536 + 1024 + nb];
  float bo = 0.0f;
  if (wg < 8) bo = b_o[wg * 16 + col];

  // ---- Recurrent state tiles live in registers (f32, C/D layout).
  f32x8 hprev0, hprev1;
#pragma unroll
  for (int e = 0; e < 8; ++e) {
    float v = hinitf[(size_t)(mbase + e + hi * 8) * HIDDEN + nbase + col];
    hprev0[e] = v;
    hprev1[e] = v;
  }

  int barcnt = 0;
  int rp = 0;  // read parity
  for (int t = 0; t < SEQ_LEN; ++t) {
    const int wp = 1 - rp;

    // ---------------- layer 0: x = X[rp], h = H0[rp] -> H0[wp] -----------
    {
      f32x8 air = {0,0,0,0,0,0,0,0}, aiz = air, ain = air;
      f32x8 ahr = air, ahz = air, ahn = air;
      const unsigned short* Xs = Xb  + (size_t)rp * BH;
      const unsigned short* Hs = H0b + (size_t)rp * BH;
#pragma unroll 4
      for (int kt = 0; kt < 16; ++kt) {
        bf16x16 ax = load_a(Xs, mbase, kt * 32, lane);
        bf16x16 ah = load_a(Hs, mbase, kt * 32, lane);
        air = wmma_bf16(ax, load_b(ldsw + (0 * 16 + kt) * 512, lane), air);
        aiz = wmma_bf16(ax, load_b(ldsw + (1 * 16 + kt) * 512, lane), aiz);
        ain = wmma_bf16(ax, load_b(ldsw + (2 * 16 + kt) * 512, lane), ain);
        ahr = wmma_bf16(ah, load_b(ldsw + (3 * 16 + kt) * 512, lane), ahr);
        ahz = wmma_bf16(ah, load_b(ldsw + (4 * 16 + kt) * 512, lane), ahz);
        ahn = wmma_bf16(ah, load_b(ldsw + (5 * 16 + kt) * 512, lane), ahn);
      }
      unsigned short* Hd = H0b + (size_t)wp * BH;
#pragma unroll
      for (int e = 0; e < 8; ++e) {
        float r = fsig(air[e] + ahr[e] + cr0);
        float z = fsig(aiz[e] + ahz[e] + cz0);
        float n = ftanh(ain[e] + bin0 + r * (ahn[e] + bhn0));
        float h = (1.0f - z) * n + z * hprev0[e];
        hprev0[e] = h;
        Hd[(size_t)(mbase + e + hi * 8) * HIDDEN + nbase + col] = f2bf(h);
      }
    }
    barcnt += NWG;
    grid_barrier(bar, barcnt);

    // ---------------- layer 1: x = H0[wp], h = H1[rp] -> H1[wp], X[wp] ----
    {
      f32x8 air = {0,0,0,0,0,0,0,0}, aiz = air, ain = air;
      f32x8 ahr = air, ahz = air, ahn = air;
      const unsigned short* Xs = H0b + (size_t)wp * BH;
      const unsigned short* Hs = H1b + (size_t)rp * BH;
#pragma unroll 4
      for (int kt = 0; kt < 16; ++kt) {
        bf16x16 ax = load_a(Xs, mbase, kt * 32, lane);
        bf16x16 ah = load_a(Hs, mbase, kt * 32, lane);
        air = wmma_bf16(ax, load_b(ldsw + (6 * 16 + kt) * 512, lane), air);
        aiz = wmma_bf16(ax, load_b(ldsw + (7 * 16 + kt) * 512, lane), aiz);
        ain = wmma_bf16(ax, load_b(ldsw + (8 * 16 + kt) * 512, lane), ain);
        ahr = wmma_bf16(ah, load_b(ldsw + (9 * 16 + kt) * 512, lane), ahr);
        ahz = wmma_bf16(ah, load_b(ldsw + (10 * 16 + kt) * 512, lane), ahz);
        ahn = wmma_bf16(ah, load_b(ldsw + (11 * 16 + kt) * 512, lane), ahn);
      }
      unsigned short* Hd = H1b + (size_t)wp * BH;
      unsigned short* Xd = Xb  + (size_t)wp * BH;
#pragma unroll
      for (int e = 0; e < 8; ++e) {
        float r = fsig(air[e] + ahr[e] + cr1);
        float z = fsig(aiz[e] + ahz[e] + cz1);
        float n = ftanh(ain[e] + bin1 + r * (ahn[e] + bhn1));
        float h = (1.0f - z) * n + z * hprev1[e];
        hprev1[e] = h;
        size_t idx = (size_t)(mbase + e + hi * 8) * HIDDEN + nbase + col;
        unsigned short hb = f2bf(h);
        Hd[idx] = hb;
        Xd[idx] = hb;
      }
    }
    barcnt += NWG;
    grid_barrier(bar, barcnt);

    // ---------------- output projection: out[:,t,:] = h1 @ W_o^T + b_o ----
    if (wg < 8) {  // uniform per block; wave w = batch tile, wg = out tile
      f32x8 acc = {0,0,0,0,0,0,0,0};
      const unsigned short* Hs = H1b + (size_t)wp * BH;
#pragma unroll 4
      for (int kt = 0; kt < 16; ++kt)
        acc = wmma_bf16(load_a(Hs, mbase, kt * 32, lane),
                        load_b(ldsw + GATE_ELEMS + kt * 512, lane), acc);
#pragma unroll
      for (int e = 0; e < 8; ++e) {
        int b = mbase + e + hi * 8;
        int o = wg * 16 + col;
        out[((size_t)b * SEQ_LEN + t) * OUTPUT + o] = acc[e] + bo;
      }
    }
    rp = wp;
  }
}

// ---------------------------------------------------------------------------
// Host-side launch.
// ---------------------------------------------------------------------------
extern "C" void kernel_launch(void* const* d_in, const int* in_sizes, int n_in,
                              void* d_out, int out_size, void* d_ws, size_t ws_size,
                              hipStream_t stream) {
  const float* z    = (const float*)d_in[0];
  const float* W_l  = (const float*)d_in[1];
  const float* b_l  = (const float*)d_in[2];
  const float* W_ih = (const float*)d_in[3];
  const float* W_hh = (const float*)d_in[4];
  const float* b_ih = (const float*)d_in[5];
  const float* b_hh = (const float*)d_in[6];
  const float* W_o  = (const float*)d_in[7];
  const float* b_o  = (const float*)d_in[8];
  float* out = (float*)d_out;

  char* ws = (char*)d_ws;
  const size_t SWZ_BYTES = (size_t)NWG * LDS_BYTES;        // 6,815,744
  unsigned short* swz  = (unsigned short*)ws;
  unsigned short* Xb   = (unsigned short*)(ws + SWZ_BYTES);
  unsigned short* H0b  = Xb  + 2 * BH;
  unsigned short* H1b  = H0b + 2 * BH;
  float*          hinitf = (float*)(H1b + 2 * BH);
  int*            bar  = (int*)(hinitf + BH);

  {
    long long total = (long long)NWG * WG_ELEMS;
    int blocks = (int)((total + 255) / 256);
    prep_weights<<<blocks, 256, 0, stream>>>(W_ih, W_hh, W_o, swz);
  }
  prep_init<<<BH / 256, 256, 0, stream>>>(z, W_l, b_l, Xb, H0b, H1b, hinitf, bar);
  gru_persistent<<<NWG, BLOCK, LDS_BYTES, stream>>>(
      swz, Xb, H0b, H1b, hinitf, b_ih, b_hh, b_o, out, bar);
}
